// Cross_transformer_70231305224244
// MI455X (gfx1250) — matrix operand order repaired and checked by reading, not compiled, not run on hardware
//
#include <hip/hip_runtime.h>
#include <math.h>

#define B_ 2
#define C_ 48
#define CP_ 64           // channel dim padded to 2 K-chunks of 32 for f16 WMMA
#define H_ 64
#define Wd_ 64
#define N_ (H_*Wd_)      // 4096 tokens, global
#define NP_ 1024         // 32*32 tokens, patch
#define EPSV 1e-5f

typedef _Float16 v16h __attribute__((ext_vector_type(16)));
typedef float    v8f  __attribute__((ext_vector_type(8)));

__device__ __constant__ int ROWC[9] = {0,0,0,16,16,16,32,32,32};
__device__ __constant__ int COLC[9] = {0,16,32,0,16,32,0,16,32};

// D(16x16 f32) = A(16x32 f16) * B(32x16 f16) + C
__device__ __forceinline__ v8f wmma16(v16h a, v16h b, v8f c) {
  return __builtin_amdgcn_wmma_f32_16x16x32_f16(false, a, false, b, (short)0, c, false, false);
}

// A-frag (16x32, 16-bit): lane m=lane&15; VGPR0-3 K=0-7 (lanes<16) / 8-15 (lanes>=16),
// VGPR4-7 K=16-23 / 24-31.  rowp -> K=0 of this row's 32-wide window (16B aligned).
__device__ __forceinline__ v16h frag_a(const _Float16* rowp, int hh) {
  union { v16h v; uint4 q[2]; } u;
  u.q[0] = *(const uint4*)(rowp + 8 * hh);
  u.q[1] = *(const uint4*)(rowp + 16 + 8 * hh);
  return u.v;
}
// B-frag (32x16, 16-bit): lane n=lane&15; VGPR0-3 K=0-7 (lanes<16) / 8-15 (lanes>=16), VGPR4-7 K=16-23 / 24-31.
__device__ __forceinline__ v16h frag_b(const _Float16* rowp, int hh) {
  union { v16h v; uint4 q[2]; } u;
  u.q[0] = *(const uint4*)(rowp + 8 * hh);
  u.q[1] = *(const uint4*)(rowp + 16 + 8 * hh);
  return u.v;
}

// ---- global projections -> split f16, token-major [pair][N][64] + channel-major [pair][48][N]
__global__ void proj_global_k(const float* __restrict__ x0, const float* __restrict__ x1,
                              const float* __restrict__ x2, const float* __restrict__ x3,
                              const float* __restrict__ w0, const float* __restrict__ w1,
                              const float* __restrict__ w2, const float* __restrict__ w3,
                              _Float16* __restrict__ gt_hi, _Float16* __restrict__ gt_lo,
                              _Float16* __restrict__ gc_hi, _Float16* __restrict__ gc_lo) {
  int t = blockIdx.x * blockDim.x + threadIdx.x;
  const int total = 4 * B_ * CP_ * N_;
  if (t >= total) return;
  int d  = t % CP_;
  int n  = (t / CP_) % N_;
  int b  = (t / (CP_ * N_)) % B_;
  int br = t / (CP_ * N_ * B_);
  const float* X  = (br == 0) ? x0 : (br == 1) ? x1 : (br == 2) ? x2 : x3;
  const float* Wm = (br == 0) ? w0 : (br == 1) ? w1 : (br == 2) ? w2 : w3;
  float acc = 0.f;
  if (d < C_) {
    #pragma unroll
    for (int c = 0; c < C_; ++c) acc += Wm[d * C_ + c] * X[(b * C_ + c) * N_ + n];
  }
  _Float16 ah = (_Float16)acc;
  _Float16 al = (_Float16)(acc - (float)ah);
  size_t pair = (size_t)br * B_ + b;
  size_t ti = (pair * N_ + n) * CP_ + d;
  gt_hi[ti] = ah; gt_lo[ti] = al;
  if (d < C_) {
    size_t ci = (pair * C_ + d) * (size_t)N_ + n;
    gc_hi[ci] = ah; gc_lo[ci] = al;
  }
}

// ---- patch projections, same dual layout per (p,br,b)
__global__ void proj_patch_k(const float* __restrict__ x0, const float* __restrict__ x1,
                             const float* __restrict__ x2, const float* __restrict__ x3,
                             const float* __restrict__ w0, const float* __restrict__ w1,
                             const float* __restrict__ w2, const float* __restrict__ w3,
                             _Float16* __restrict__ pt_hi, _Float16* __restrict__ pt_lo,
                             _Float16* __restrict__ pc_hi, _Float16* __restrict__ pc_lo) {
  int t = blockIdx.x * blockDim.x + threadIdx.x;
  const int total = 9 * 4 * B_ * CP_ * NP_;
  if (t >= total) return;
  int d  = t % CP_;
  int np = (t / CP_) % NP_;
  int b  = (t / (CP_ * NP_)) % B_;
  int br = (t / (CP_ * NP_ * B_)) % 4;
  int p  = t / (CP_ * NP_ * B_ * 4);
  const float* X  = (br == 0) ? x0 : (br == 1) ? x1 : (br == 2) ? x2 : x3;
  const float* Wm = (br == 0) ? w0 : (br == 1) ? w1 : (br == 2) ? w2 : w3;
  int gi = ROWC[p] + (np >> 5);
  int gj = COLC[p] + (np & 31);
  float acc = 0.f;
  if (d < C_) {
    #pragma unroll
    for (int c = 0; c < C_; ++c)
      acc += Wm[p * C_ * C_ + d * C_ + c] * X[(b * C_ + c) * N_ + gi * Wd_ + gj];
  }
  _Float16 ah = (_Float16)acc;
  _Float16 al = (_Float16)(acc - (float)ah);
  size_t pair = (size_t)(p * 4 + br) * B_ + b;
  size_t ti = (pair * NP_ + np) * CP_ + d;
  pt_hi[ti] = ah; pt_lo[ti] = al;
  if (d < C_) {
    size_t ci = (pair * C_ + d) * (size_t)NP_ + np;
    pc_hi[ci] = ah; pc_lo[ci] = al;
  }
}

// ---- pass 1: rZ[n] = 1 / sum_m exp(-e[n,m]); e via split-f16 WMMA (hi*hi + hi*lo + lo*hi)
__global__ void __launch_bounds__(256) attn_z_k(const _Float16* __restrict__ th,
                                                const _Float16* __restrict__ tl,
                                                float* __restrict__ rZ, int mode) {
  const int lane = threadIdx.x & 31;
  const int wv   = threadIdx.x >> 5;
  const int pair = blockIdx.y;
  const int Ntok = mode ? NP_ : N_;
  int rb = blockIdx.x * 8 + wv;
  if (rb >= (Ntok >> 4)) return;                // wave-uniform: EXEC stays all-ones
  const int n0 = rb << 4;
  size_t qp, kp;
  if (mode == 0) { int b = pair >> 2, br = pair & 3; qp = b; kp = (size_t)br * B_ + b; }
  else { int p = pair >> 3, b = (pair >> 2) & 1, br = pair & 3;
         qp = (size_t)(p * 4) * B_ + b; kp = (size_t)(p * 4 + br) * B_ + b; }
  const _Float16* Qh = th + qp * (size_t)Ntok * CP_;
  const _Float16* Ql = tl + qp * (size_t)Ntok * CP_;
  const _Float16* Kh = th + kp * (size_t)Ntok * CP_;
  const _Float16* Kl = tl + kp * (size_t)Ntok * CP_;
  float* rz = rZ + (size_t)pair * Ntok;
  const int row = lane & 15;
  const int hh  = lane >> 4;
  size_t qrow = (size_t)(n0 + row) * CP_;
  v16h aQ[4];                                   // [chunk][hi/lo]
  aQ[0] = frag_a(Qh + qrow + 0,  hh);
  aQ[1] = frag_a(Ql + qrow + 0,  hh);
  aQ[2] = frag_a(Qh + qrow + 32, hh);
  aQ[3] = frag_a(Ql + qrow + 32, hh);
  float zp[8] = {0.f,0.f,0.f,0.f,0.f,0.f,0.f,0.f};
  for (int m0 = 0; m0 < Ntok; m0 += 16) {
    size_t mrow = (size_t)(m0 + row) * CP_;
    v8f acc = {0.f,0.f,0.f,0.f,0.f,0.f,0.f,0.f};
    #pragma unroll
    for (int k = 0; k < 2; ++k) {
      v16h bh = frag_b(Kh + mrow + 32 * k, hh);
      v16h bl = frag_b(Kl + mrow + 32 * k, hh);
      acc = wmma16(aQ[2 * k],     bh, acc);
      acc = wmma16(aQ[2 * k],     bl, acc);
      acc = wmma16(aQ[2 * k + 1], bh, acc);
    }
    #pragma unroll
    for (int g = 0; g < 8; ++g) zp[g] += __expf(-acc[g]);
  }
  #pragma unroll
  for (int g = 0; g < 8; ++g) {
    float s = zp[g];
    s += __shfl_xor(s, 8, 16);
    s += __shfl_xor(s, 4, 16);
    s += __shfl_xor(s, 2, 16);
    s += __shfl_xor(s, 1, 16);
    if (row == 0) rz[n0 + g + 8 * hh] = 1.0f / s;
  }
}

// ---- pass 2: out[c,m] = sum_n V[c,n]*exp(-e[n,m])*rZ[n]; both GEMMs split-f16 WMMA
__global__ void __launch_bounds__(256) attn_av_k(const _Float16* __restrict__ th,
                                                 const _Float16* __restrict__ tl,
                                                 const _Float16* __restrict__ ch,
                                                 const _Float16* __restrict__ cl,
                                                 const float* __restrict__ rZ,
                                                 float* __restrict__ cat, int mode) {
  __shared__ __align__(16) _Float16 lds[8 * 1280];   // per wave: hi[16][40] + lo[16][40]
  const int lane = threadIdx.x & 31;
  const int wv   = threadIdx.x >> 5;
  const int pair = blockIdx.y;
  const int Ntok = mode ? NP_ : N_;
  const int wcount = 3 * (Ntok >> 4);
  int w = blockIdx.x * 8 + wv;
  if (w >= wcount) return;
  const int c0 = (w % 3) * 16;
  const int m0 = (w / 3) * 16;
  size_t qp, kp; float* co;
  if (mode == 0) {
    int b = pair >> 2, br = pair & 3;
    qp = b; kp = (size_t)br * B_ + b;
    co = cat + (size_t)(b * 4 * C_ + br * C_) * N_;
  } else {
    int p = pair >> 3, b = (pair >> 2) & 1, br = pair & 3;
    qp = (size_t)(p * 4) * B_ + b; kp = (size_t)(p * 4 + br) * B_ + b;
    co = cat + (size_t)((p * B_ + b) * 4 * C_ + br * C_) * NP_;
  }
  const _Float16* Qh = th + qp * (size_t)Ntok * CP_;
  const _Float16* Ql = tl + qp * (size_t)Ntok * CP_;
  const _Float16* Kh = th + kp * (size_t)Ntok * CP_;
  const _Float16* Kl = tl + kp * (size_t)Ntok * CP_;
  const _Float16* Vh = ch + kp * (size_t)C_ * Ntok;  // channel-major
  const _Float16* Vl = cl + kp * (size_t)C_ * Ntok;
  const float* rz = rZ + (size_t)pair * Ntok;
  const int row = lane & 15;
  const int hh  = lane >> 4;
  _Float16* hiW = lds + wv * 1280;
  _Float16* loW = hiW + 640;
  size_t mrow = (size_t)(m0 + row) * CP_;
  v16h bK[4];                                   // K-columns at m0, invariant over n loop
  bK[0] = frag_b(Kh + mrow + 0,  hh);
  bK[1] = frag_b(Kl + mrow + 0,  hh);
  bK[2] = frag_b(Kh + mrow + 32, hh);
  bK[3] = frag_b(Kl + mrow + 32, hh);
  v8f out = {0.f,0.f,0.f,0.f,0.f,0.f,0.f,0.f};
  for (int n0 = 0; n0 < Ntok; n0 += 32) {
    #pragma unroll
    for (int sub = 0; sub < 2; ++sub) {
      int nt = n0 + 16 * sub;
      size_t qrow = (size_t)(nt + row) * CP_;
      v8f e = {0.f,0.f,0.f,0.f,0.f,0.f,0.f,0.f};
      #pragma unroll
      for (int k = 0; k < 2; ++k) {
        v16h ah = frag_a(Qh + qrow + 32 * k, hh);
        v16h al = frag_a(Ql + qrow + 32 * k, hh);
        e = wmma16(ah, bK[2 * k],     e);
        e = wmma16(ah, bK[2 * k + 1], e);
        e = wmma16(al, bK[2 * k],     e);
      }
      #pragma unroll
      for (int g = 0; g < 8; g += 2) {          // C-layout -> split a, stored [m][n] (n contiguous) in LDS
        int nl = 16 * sub + g + 8 * hh;
        float a0 = __expf(-e[g])     * rz[n0 + nl];
        float a1 = __expf(-e[g + 1]) * rz[n0 + nl + 1];
        union { uint u; _Float16 hx[2]; } ph, pl;
        ph.hx[0] = (_Float16)a0;
        ph.hx[1] = (_Float16)a1;
        pl.hx[0] = (_Float16)(a0 - (float)ph.hx[0]);
        pl.hx[1] = (_Float16)(a1 - (float)ph.hx[1]);
        *(uint*)(hiW + row * 40 + nl) = ph.u;
        *(uint*)(loW + row * 40 + nl) = pl.u;
      }
    }
    size_t vrow = (size_t)(c0 + row) * Ntok + n0;
    v16h vh  = frag_a(Vh + vrow, hh);
    v16h vl  = frag_a(Vl + vrow, hh);
    v16h ath = frag_b(hiW + row * 40, hh);
    v16h atl = frag_b(loW + row * 40, hh);
    out = wmma16(vh, ath, out);
    out = wmma16(vh, atl, out);
    out = wmma16(vl, ath, out);
  }
  #pragma unroll
  for (int g = 0; g < 8; ++g)
    co[(size_t)(c0 + g + 8 * hh) * Ntok + m0 + row] = out[g];
}

// ---- fuse (global): 1x1 conv 4C->C + BN + ReLU, residual + gamma
__global__ void fuse_global_k(const float* __restrict__ catG, const float* __restrict__ fw,
                              const float* __restrict__ fb, const float* __restrict__ fg,
                              const float* __restrict__ fbb, const float* __restrict__ fm,
                              const float* __restrict__ fv, const float* __restrict__ gamma_all,
                              const float* __restrict__ input, float* __restrict__ out0) {
  int t = blockIdx.x * blockDim.x + threadIdx.x;
  if (t >= B_ * C_ * N_) return;
  int n = t % N_;
  int o = (t / N_) % C_;
  int b = t / (N_ * C_);
  float acc = fb[o];
  #pragma unroll 4
  for (int c4 = 0; c4 < 4 * C_; ++c4)
    acc += fw[o * 4 * C_ + c4] * catG[((size_t)b * 4 * C_ + c4) * N_ + n];
  float y = (acc - fm[o]) * (fg[o] / sqrtf(fv[o] + EPSV)) + fbb[o];
  y = fmaxf(y, 0.f);
  out0[t] = input[t] + gamma_all[0] * y;
}

// ---- fuse (patch): per-patch 1x1 conv + BN + ReLU, scaled
__global__ void fuse_patch_k(const float* __restrict__ catP, const float* __restrict__ fpw,
                             const float* __restrict__ fpb, const float* __restrict__ fpg,
                             const float* __restrict__ fpbb, const float* __restrict__ fpm,
                             const float* __restrict__ fpv, const float* __restrict__ gp,
                             float* __restrict__ patchOut) {
  int t = blockIdx.x * blockDim.x + threadIdx.x;
  if (t >= 9 * B_ * C_ * NP_) return;
  int np = t % NP_;
  int o  = (t / NP_) % C_;
  int b  = (t / (NP_ * C_)) % B_;
  int p  = t / (NP_ * C_ * B_);
  float acc = fpb[p * C_ + o];
  #pragma unroll 4
  for (int c4 = 0; c4 < 4 * C_; ++c4)
    acc += fpw[((size_t)p * C_ + o) * 4 * C_ + c4] * catP[(((size_t)p * B_ + b) * 4 * C_ + c4) * NP_ + np];
  float y = (acc - fpm[p * C_ + o]) * (fpg[p * C_ + o] / sqrtf(fpv[p * C_ + o] + EPSV)) + fpbb[p * C_ + o];
  patchOut[t] = 0.5f * gp[p] * fmaxf(y, 0.f);
}

// ---- accumulate overlapping patches (gather, deterministic)
__global__ void combine_k(float* __restrict__ out0, const float* __restrict__ patchOut) {
  int t = blockIdx.x * blockDim.x + threadIdx.x;
  if (t >= B_ * C_ * N_) return;
  int n = t % N_;
  int o = (t / N_) % C_;
  int b = t / (N_ * C_);
  int h = n >> 6, w = n & 63;
  float v = out0[t];
  #pragma unroll
  for (int p = 0; p < 9; ++p) {
    int i = h - ROWC[p], j = w - COLC[p];
    if (i >= 0 && i < 32 && j >= 0 && j < 32)
      v += patchOut[(((size_t)p * B_ + b) * C_ + o) * NP_ + i * 32 + j];
  }
  out0[t] = v;
}

// ---- to_out: 3x3 SAME conv + BN + ReLU
__global__ void conv_k(const float* __restrict__ out0, const float* __restrict__ ow,
                       const float* __restrict__ ob, const float* __restrict__ og,
                       const float* __restrict__ obb, const float* __restrict__ om,
                       const float* __restrict__ ov, float* __restrict__ dst) {
  int t = blockIdx.x * blockDim.x + threadIdx.x;
  if (t >= B_ * C_ * N_) return;
  int n = t % N_;
  int o = (t / N_) % C_;
  int b = t / (N_ * C_);
  int h = n >> 6, w = n & 63;
  float acc = ob[o];
  for (int c = 0; c < C_; ++c) {
    const float* src = out0 + ((size_t)b * C_ + c) * N_;
    const float* wk  = ow + ((size_t)o * C_ + c) * 9;
    #pragma unroll
    for (int dy = 0; dy < 3; ++dy) {
      int hhh = h + dy - 1;
      if (hhh < 0 || hhh >= H_) continue;
      #pragma unroll
      for (int dx = 0; dx < 3; ++dx) {
        int www = w + dx - 1;
        if (www < 0 || www >= Wd_) continue;
        acc += wk[dy * 3 + dx] * src[hhh * Wd_ + www];
      }
    }
  }
  float y = (acc - om[o]) * (og[o] / sqrtf(ov[o] + EPSV)) + obb[o];
  dst[t] = fmaxf(y, 0.f);
}

extern "C" void kernel_launch(void* const* d_in, const int* in_sizes, int n_in,
                              void* d_out, int out_size, void* d_ws, size_t ws_size,
                              hipStream_t stream) {
  (void)in_sizes; (void)n_in; (void)out_size; (void)ws_size;
  const float* input   = (const float*)d_in[0];
  const float* feat_b  = (const float*)d_in[1];
  const float* feat_c  = (const float*)d_in[2];
  const float* feat_d  = (const float*)d_in[3];
  const float* Wfa_all = (const float*)d_in[4];
  const float* Wfb_all = (const float*)d_in[5];
  const float* Wfc_all = (const float*)d_in[6];
  const float* Wfd_all = (const float*)d_in[7];
  const float* Wfa     = (const float*)d_in[8];
  const float* Wfb     = (const float*)d_in[9];
  const float* Wfc     = (const float*)d_in[10];
  const float* Wfd     = (const float*)d_in[11];
  const float* fuse_w  = (const float*)d_in[12];
  const float* fuse_b  = (const float*)d_in[13];
  const float* fuse_g  = (const float*)d_in[14];
  const float* fuse_bb = (const float*)d_in[15];
  const float* fuse_m  = (const float*)d_in[16];
  const float* fuse_v  = (const float*)d_in[17];
  const float* fuseP_w = (const float*)d_in[18];
  const float* fuseP_b = (const float*)d_in[19];
  const float* fuseP_g = (const float*)d_in[20];
  const float* fuseP_bb= (const float*)d_in[21];
  const float* fuseP_m = (const float*)d_in[22];
  const float* fuseP_v = (const float*)d_in[23];
  const float* gamma_all = (const float*)d_in[24];
  const float* gamma_p = (const float*)d_in[25];
  const float* out_w   = (const float*)d_in[26];
  const float* out_b   = (const float*)d_in[27];
  const float* out_g   = (const float*)d_in[28];
  const float* out_bb  = (const float*)d_in[29];
  const float* out_m   = (const float*)d_in[30];
  const float* out_v   = (const float*)d_in[31];

  // float scratch first, then 16B-aligned f16 scratch
  float* wsf = (float*)d_ws;
  size_t off = 0;
  float* catG     = wsf + off; off += (size_t)B_ * 4 * C_ * N_;     // 1,572,864
  float* out0     = wsf + off; off += (size_t)B_ * C_ * N_;         //   393,216
  float* catP     = wsf + off; off += (size_t)9 * B_ * 4 * C_ * NP_;// 3,538,944
  float* patchOut = wsf + off; off += (size_t)9 * B_ * C_ * NP_;    //   884,736
  float* rZg      = wsf + off; off += (size_t)8 * N_;               //    32,768
  float* rZp      = wsf + off; off += (size_t)72 * NP_;             //    73,728
  _Float16* wsh = (_Float16*)(wsf + off);
  size_t hoff = 0;
  _Float16* gt_hi = wsh + hoff; hoff += (size_t)4 * B_ * N_ * CP_;      // 2,097,152
  _Float16* gt_lo = wsh + hoff; hoff += (size_t)4 * B_ * N_ * CP_;
  _Float16* gc_hi = wsh + hoff; hoff += (size_t)4 * B_ * C_ * N_;       // 1,572,864
  _Float16* gc_lo = wsh + hoff; hoff += (size_t)4 * B_ * C_ * N_;
  _Float16* pt_hi = wsh + hoff; hoff += (size_t)9 * 4 * B_ * NP_ * CP_; // 4,718,592
  _Float16* pt_lo = wsh + hoff; hoff += (size_t)9 * 4 * B_ * NP_ * CP_;
  _Float16* pc_hi = wsh + hoff; hoff += (size_t)9 * 4 * B_ * C_ * NP_;  // 3,538,944
  _Float16* pc_lo = wsh + hoff; hoff += (size_t)9 * 4 * B_ * C_ * NP_;  // ~74 MB total

  // global branch
  proj_global_k<<<(4 * B_ * CP_ * N_ + 255) / 256, 256, 0, stream>>>(
      input, feat_b, feat_c, feat_d, Wfa_all, Wfb_all, Wfc_all, Wfd_all,
      gt_hi, gt_lo, gc_hi, gc_lo);
  attn_z_k<<<dim3(32, 8), 256, 0, stream>>>(gt_hi, gt_lo, rZg, 0);
  attn_av_k<<<dim3(96, 8), 256, 0, stream>>>(gt_hi, gt_lo, gc_hi, gc_lo, rZg, catG, 0);
  fuse_global_k<<<(B_ * C_ * N_ + 255) / 256, 256, 0, stream>>>(
      catG, fuse_w, fuse_b, fuse_g, fuse_bb, fuse_m, fuse_v, gamma_all, input, out0);

  // patch branches
  proj_patch_k<<<(9 * 4 * B_ * CP_ * NP_ + 255) / 256, 256, 0, stream>>>(
      input, feat_b, feat_c, feat_d, Wfa, Wfb, Wfc, Wfd,
      pt_hi, pt_lo, pc_hi, pc_lo);
  attn_z_k<<<dim3(8, 72), 256, 0, stream>>>(pt_hi, pt_lo, rZp, 1);
  attn_av_k<<<dim3(24, 72), 256, 0, stream>>>(pt_hi, pt_lo, pc_hi, pc_lo, rZp, catP, 1);
  fuse_patch_k<<<(9 * B_ * C_ * NP_ + 255) / 256, 256, 0, stream>>>(
      catP, fuseP_w, fuseP_b, fuseP_g, fuseP_bb, fuseP_m, fuseP_v, gamma_p, patchOut);
  combine_k<<<(B_ * C_ * N_ + 255) / 256, 256, 0, stream>>>(out0, patchOut);

  // to_out
  conv_k<<<(B_ * C_ * N_ + 255) / 256, 256, 0, stream>>>(
      out0, out_w, out_b, out_g, out_bb, out_m, out_v, (float*)d_out);
}
